// PNASimplifiedLayer_8418135900210
// MI455X (gfx1250) — compile-verified, matrix operand order checked by reading, not run on hardware
//
#include <hip/hip_runtime.h>
#include <hip/hip_bf16.h>
#include <math.h>

#define FDIM 128
#define KTOT (13 * FDIM)      // 1664
#define KPAD (KTOT + 8)       // LDS row pad: stride 3344B -> 4-bank skew, conflict-free
#define ROWS 32               // nodes per block (two 16-row WMMA tiles)
#define AVG_D_LOG 2.6f
#define PNA_EPS 1e-5f

typedef __attribute__((ext_vector_type(16))) __bf16 v16bf;
typedef __attribute__((ext_vector_type(8)))  __bf16 v8bf;
typedef __attribute__((ext_vector_type(8)))  float  v8f;

// Monotonic float<->int bit tricks for lock-free float min/max atomics.
__device__ __forceinline__ void atomicMaxF(float* addr, float v) {
    if (v >= 0.0f) atomicMax((int*)addr, __float_as_int(v));
    else           atomicMin((unsigned int*)addr, __float_as_uint(v));
}
__device__ __forceinline__ void atomicMinF(float* addr, float v) {
    if (v >= 0.0f) atomicMin((int*)addr, __float_as_int(v));
    else           atomicMax((unsigned int*)addr, __float_as_uint(v));
}

// ---------------- init workspace (ws is poisoned; must re-init every call) ---
__global__ void k_init(float* deg, float* ssum, float* ssq, float* smx, float* smn,
                       float* colsum, float* colsq, int n, long total) {
    long i = (long)blockIdx.x * blockDim.x + threadIdx.x;
    if (i < total) {
        ssum[i] = 0.0f;
        ssq[i]  = 0.0f;
        smx[i]  = -__builtin_inff();
        smn[i]  =  __builtin_inff();
    }
    if (i < n) deg[i] = 0.0f;
    if (i < FDIM) { colsum[i] = 0.0f; colsq[i] = 0.0f; }
}

// ---------------- pack W (f32 row-major 1664x128) into bf16 WMMA-B fragment order
// B frag (32x16 bf16, wave32): lane = col + 16*((k>>3)&1); slot = (k&7) + 8*(k>=16)
__global__ void k_packW(const float* __restrict__ W, __bf16* __restrict__ Wp) {
    int i = blockIdx.x * blockDim.x + threadIdx.x;
    if (i >= KTOT * FDIM) return;
    int k = i / FDIM, nn = i % FDIM;
    int kc = k >> 5, kk = k & 31, ct = nn >> 4, col = nn & 15;
    int grp  = kk >> 3;
    int lane = col + ((grp & 1) << 4);
    int slot = (kk & 7) + ((grp >> 1) << 3);
    Wp[(((size_t)(kc * 8 + ct)) * 32 + lane) * 16 + slot] = (__bf16)W[i];
}

// ---------------- edge scatter: one wave per edge, 4 features per lane --------
__global__ void k_edges(const float* __restrict__ h, const int* __restrict__ src,
                        const int* __restrict__ dst, float* deg,
                        float* ssum, float* ssq, float* smx, float* smn, int E) {
    int tid = blockIdx.x * blockDim.x + threadIdx.x;
    int e = tid >> 5;
    if (e >= E) return;
    int lane = tid & 31;
    int s = src[e], d = dst[e];
    if (lane == 0) atomicAdd(&deg[d], 1.0f);
    const float4 v4 = *reinterpret_cast<const float4*>(h + (size_t)s * FDIM + lane * 4);
    size_t off = (size_t)d * FDIM + lane * 4;
    float vv[4] = {v4.x, v4.y, v4.z, v4.w};
#pragma unroll
    for (int c = 0; c < 4; ++c) {
        float v = vv[c];
        atomicAdd(&ssum[off + c], v);
        atomicAdd(&ssq[off + c], v * v);
        atomicMaxF(&smx[off + c], v);
        atomicMinF(&smn[off + c], v);
    }
}

// ---------------- fused: stats -> hc tile in LDS (bf16) -> WMMA GEMM -> ReLU
//                  -> store pre-BN + per-column BN partial sums ----------------
// 32 nodes per block: two 16-row tiles share each B fragment (half the L2 W
// traffic) and give every wave two independent WMMA accumulation chains.
__global__ __launch_bounds__(256)
void k_gemm(const float* __restrict__ h, const float* __restrict__ deg,
            const float* __restrict__ ssum, const float* __restrict__ ssq,
            const float* __restrict__ smx, const float* __restrict__ smn,
            const __bf16* __restrict__ Wp, const float* __restrict__ bias,
            float* __restrict__ pre, float* colsum, float* colsq, int n) {
    __shared__ __bf16 Atile[ROWS * KPAD];   // 32 x 1672 bf16 = ~104 KB (320 KB/WGP on gfx1250)
    const int tid   = threadIdx.x;
    const int rtile = blockIdx.x;

    // Phase 1: build hc tile. thread -> (row = tid>>3, 16 features)
    {
        int lrow = tid >> 3;                       // 0..31
        int node = rtile * ROWS + lrow;
        if (node > n - 1) node = n - 1;            // clamp (WMMA needs full tiles)
        int f0 = (tid & 7) * 16;
        float dg   = fmaxf(deg[node], 1.0f);
        float inv  = 1.0f / dg;
        float logd = logf(dg + 1.0f);
        float sc2  = logd * (1.0f / AVG_D_LOG);
        float sc3  = AVG_D_LOG / logd;
        __bf16* Ar = &Atile[lrow * KPAD];
        size_t nb = (size_t)node * FDIM;
#pragma unroll
        for (int j = 0; j < 16; ++j) {
            int f = f0 + j;
            float hv   = h[nb + f];
            float mean = ssum[nb + f] * inv;
            float mx   = smx[nb + f];
            float mn   = smn[nb + f];
            float var  = ssq[nb + f] * inv - mean * mean;
            float sd   = sqrtf(fmaxf(var, 0.0f) + PNA_EPS);
            Ar[f] = (__bf16)hv;
            float a[4] = {mean, mx, mn, sd};
#pragma unroll
            for (int q = 0; q < 4; ++q) {
                Ar[FDIM + q * FDIM + f]       = (__bf16)(a[q]);
                Ar[FDIM + (4 + q) * FDIM + f] = (__bf16)(a[q] * sc2);
                Ar[FDIM + (8 + q) * FDIM + f] = (__bf16)(a[q] * sc3);
            }
        }
    }
    __syncthreads();

    // Phase 2: wave -> col tile; two row halves per wave, one shared B frag.
    const int wave  = tid >> 5;        // col tile 0..7
    const int lane  = tid & 31;
    const int m     = lane & 15;       // A row within 16-row tile
    const int khalf = (lane < 16) ? 0 : 8;
    v8f acc0 = {};
    v8f acc1 = {};
    const v8bf*  Arow0 = reinterpret_cast<const v8bf*>(Atile + m * KPAD);
    const v8bf*  Arow1 = reinterpret_cast<const v8bf*>(Atile + (m + 16) * KPAD);
    const v16bf* Bp    = reinterpret_cast<const v16bf*>(Wp);
#pragma unroll 2
    for (int kc = 0; kc < KTOT / 32; ++kc) {
        // 16-bit A 16x32 lane map: lane<16 holds K {0..7,16..23}, lane>=16 {8..15,24..31}
        int i0 = (kc * 32 + khalf) >> 3;
        int i1 = (kc * 32 + khalf + 16) >> 3;
        v16bf bfrag = Bp[(size_t)(kc * 8 + wave) * 32 + lane];
        v8bf lo0 = Arow0[i0], hi0 = Arow0[i1];
        v8bf lo1 = Arow1[i0], hi1 = Arow1[i1];
        v16bf a0 = __builtin_shufflevector(lo0, hi0,
            0, 1, 2, 3, 4, 5, 6, 7, 8, 9, 10, 11, 12, 13, 14, 15);
        v16bf a1 = __builtin_shufflevector(lo1, hi1,
            0, 1, 2, 3, 4, 5, 6, 7, 8, 9, 10, 11, 12, 13, 14, 15);
        acc0 = __builtin_amdgcn_wmma_f32_16x16x32_bf16(
            false, a0, false, bfrag, (short)0, acc0, false, false);
        acc1 = __builtin_amdgcn_wmma_f32_16x16x32_bf16(
            false, a1, false, bfrag, (short)0, acc1, false, false);
    }

    // Phase 3: bias + ReLU, store + column BN partials.
    // Block-uniform full/tail split: only the last block can have invalid rows,
    // so the common path has no per-lane EXEC masking around its stores.
    int col = wave * 16 + m;
    float bc = bias[col];
    float s8 = 0.0f, q8 = 0.0f;
    int mbase = (lane < 16) ? 0 : 8;   // D layout: VGPR i -> M = i (+8 for hi lanes)
    long rowbase = (long)rtile * ROWS;
    if (rowbase + ROWS <= (long)n) {
        float* p0 = pre + (rowbase + mbase) * FDIM + col;
#pragma unroll
        for (int i = 0; i < 8; ++i) {
            float v0 = fmaxf(acc0[i] + bc, 0.0f);
            float v1 = fmaxf(acc1[i] + bc, 0.0f);
            p0[(size_t)i * FDIM] = v0;
            p0[(size_t)(i + 16) * FDIM] = v1;
            s8 += v0 + v1;
            q8 += v0 * v0 + v1 * v1;
        }
    } else {
#pragma unroll
        for (int i = 0; i < 8; ++i) {
            long r0 = rowbase + mbase + i;
            float v0 = fmaxf(acc0[i] + bc, 0.0f);
            if (r0 < n) {
                pre[r0 * FDIM + col] = v0;
                s8 += v0; q8 += v0 * v0;
            }
            long r1 = rowbase + 16 + mbase + i;
            float v1 = fmaxf(acc1[i] + bc, 0.0f);
            if (r1 < n) {
                pre[r1 * FDIM + col] = v1;
                s8 += v1; q8 += v1 * v1;
            }
        }
    }
    // lanes L and L+16 share a column -> combine, then one atomic per column.
    s8 += __shfl_xor(s8, 16, 32);
    q8 += __shfl_xor(q8, 16, 32);
    if (lane < 16) {
        atomicAdd(&colsum[col], s8);
        atomicAdd(&colsq[col], q8);
    }
}

// ---------------- BatchNorm finalize ----------------------------------------
__global__ void k_bnprep(const float* colsum, const float* colsq,
                         const float* gma, const float* bta,
                         float* scale, float* shift, float invN) {
    int c = threadIdx.x;
    if (c < FDIM) {
        float mu  = colsum[c] * invN;
        float var = fmaxf(colsq[c] * invN - mu * mu, 0.0f);
        float s   = gma[c] * rsqrtf(var + PNA_EPS);
        scale[c] = s;
        shift[c] = bta[c] - mu * s;
    }
}

__global__ void k_bnapply(const float* __restrict__ pre, const float* __restrict__ scale,
                          const float* __restrict__ shift, float* __restrict__ out, long total) {
    long i = (long)blockIdx.x * blockDim.x + threadIdx.x;
    if (i < total) {
        int c = (int)(i & (FDIM - 1));
        out[i] = pre[i] * scale[c] + shift[c];
    }
}

extern "C" void kernel_launch(void* const* d_in, const int* in_sizes, int n_in,
                              void* d_out, int out_size, void* d_ws, size_t ws_size,
                              hipStream_t stream) {
    const float* h     = (const float*)d_in[0];
    const int*   src   = (const int*)d_in[1];
    const int*   dst   = (const int*)d_in[2];
    const float* W     = (const float*)d_in[3];
    const float* bias  = (const float*)d_in[4];
    const float* gma   = (const float*)d_in[5];
    const float* bta   = (const float*)d_in[6];
    float* out = (float*)d_out;

    int n = in_sizes[0] / FDIM;   // 50000
    int E = in_sizes[1];          // ~650000

    // workspace carve-up (all f32 except packed bf16 W at the end)
    float* ws     = (float*)d_ws;
    float* deg    = ws;
    float* ssum   = deg + n;
    float* ssq    = ssum + (size_t)n * FDIM;
    float* smx    = ssq  + (size_t)n * FDIM;
    float* smn    = smx  + (size_t)n * FDIM;
    float* colsum = smn  + (size_t)n * FDIM;
    float* colsq  = colsum + FDIM;
    float* scale  = colsq  + FDIM;
    float* shift  = scale  + FDIM;
    float* pre    = shift  + FDIM;
    __bf16* Wp    = (__bf16*)(pre + (size_t)n * FDIM);

    long total = (long)n * FDIM;
    int nblk   = (int)((total + 255) / 256);

    k_init<<<nblk, 256, 0, stream>>>(deg, ssum, ssq, smx, smn, colsum, colsq, n, total);
    k_packW<<<(KTOT * FDIM + 255) / 256, 256, 0, stream>>>(W, Wp);
    k_edges<<<(E + 7) / 8, 256, 0, stream>>>(h, src, dst, deg, ssum, ssq, smx, smn, E);
    k_gemm<<<(n + ROWS - 1) / ROWS, 256, 0, stream>>>(h, deg, ssum, ssq, smx, smn, Wp, bias,
                                                      pre, colsum, colsq, n);
    k_bnprep<<<1, FDIM, 0, stream>>>(colsum, colsq, gma, bta, scale, shift, 1.0f / (float)n);
    k_bnapply<<<nblk, 256, 0, stream>>>(pre, scale, shift, out, total);
}
